// HypER_39565238731233
// MI455X (gfx1250) — matrix operand order repaired and checked by reading, not compiled, not run on hardware
//
#include <hip/hip_runtime.h>

typedef __attribute__((ext_vector_type(16))) __bf16 v16bf;
typedef __attribute__((ext_vector_type(8)))  __bf16 v8bf;
typedef __attribute__((ext_vector_type(8)))  float  v8f;
typedef __attribute__((ext_vector_type(4)))  int    v4i;

#define B_    1024
#define DENT  400
#define OUTC  32
#define KW_   9
#define L_    392            // 400 - 9 + 1
#define HID   (OUTC * L_)    // 12544
#define NENT  50000
#define KP    416            // 400 padded to 13*32 for final GEMM
#define EPS_  1e-5f

#if __has_builtin(__builtin_amdgcn_global_load_async_to_lds_b128) && \
    __has_builtin(__builtin_amdgcn_s_wait_asynccnt)
#define HAVE_ASYNC_LDS 1
#else
#define HAVE_ASYNC_LDS 0
#endif

// ---- workspace layout (bytes), everything 256B aligned ----
#define WS_STATS    0u                 // 1024 floats: [0]=s0 [1]=ss0 [2..33]=bn1 sum [34..65]=bn1 ssq
                                       // [66..465]=mean2 [466..865]=rstd2
#define WS_KBUF     4096u                          // f32 [1024][288]
#define WS_HBF      (WS_KBUF  + 1179648u)          // u16 [1024][12544]
#define WS_FCWBF    (WS_HBF   + 25690112u)         // u16 [400][12544]
#define WS_HF32     (WS_FCWBF + 10035200u)         // f32 [1024][400]
#define WS_HRBF     (WS_HF32  + 1638400u)          // u16 [1024][416]
#define WS_ENTBF    (WS_HRBF  + 851968u)           // u16 [50000][416]

__device__ __forceinline__ unsigned short f2bf(float f) {
  unsigned int u = __float_as_uint(f);
  u += 0x7FFFu + ((u >> 16) & 1u);     // round-to-nearest-even
  return (unsigned short)(u >> 16);
}

// ---------------- 0: zero stats region ----------------
__global__ __launch_bounds__(256) void k_zero_stats(float* stats) {
  for (int i = threadIdx.x; i < 1024; i += 256) stats[i] = 0.f;
}

// ---------------- 1: bn0 global sum/sumsq over e1_emb ----------------
__global__ __launch_bounds__(256) void k_bn0_stats(const float* __restrict__ x, float* stats) {
  float s = 0.f, ss = 0.f;
  int idx = blockIdx.x * 256 + threadIdx.x;
  for (int i = idx; i < B_ * DENT; i += gridDim.x * 256) {
    float v = x[i]; s += v; ss += v * v;
  }
  __shared__ float sh[256], sh2[256];
  sh[threadIdx.x] = s; sh2[threadIdx.x] = ss; __syncthreads();
  for (int st = 128; st > 0; st >>= 1) {
    if (threadIdx.x < st) { sh[threadIdx.x] += sh[threadIdx.x + st]; sh2[threadIdx.x] += sh2[threadIdx.x + st]; }
    __syncthreads();
  }
  if (threadIdx.x == 0) { atomicAdd(&stats[0], sh[0]); atomicAdd(&stats[1], sh2[0]); }
}

// ---------------- 2: hypernetwork k = rel @ fc1_w.T + fc1_b ----------------
__global__ __launch_bounds__(256) void k_hyper(const float* __restrict__ rel,
                                               const float* __restrict__ w,
                                               const float* __restrict__ bias,
                                               float* __restrict__ kbuf) {
  int idx = blockIdx.x * 256 + threadIdx.x;
  if (idx >= B_ * OUTC * KW_) return;
  int b = idx / (OUTC * KW_), j = idx % (OUTC * KW_);
  const float* r = rel + b * DENT;
  const float* wr = w + j * DENT;
  float acc = bias[j];
  for (int k = 0; k < DENT; ++k) acc = fmaf(r[k], wr[k], acc);
  kbuf[idx] = acc;
}

// ---------------- 3/4: per-sample conv (stats pass / normalize pass) ----------------
template <int PASS>
__global__ __launch_bounds__(128) void k_conv(const float* __restrict__ x,
                                              const float* __restrict__ kbuf,
                                              const float* __restrict__ g0p, const float* __restrict__ b0p,
                                              const float* __restrict__ g1p, const float* __restrict__ b1p,
                                              float* stats, unsigned short* __restrict__ hbf) {
  int b = blockIdx.x >> 5;
  int o = blockIdx.x & 31;
  const float inv0 = 1.0f / (float)(B_ * DENT);
  float mu0 = stats[0] * inv0;
  float var0 = stats[1] * inv0 - mu0 * mu0;
  float rs0 = rsqrtf(var0 + EPS_);
  float g0 = g0p[0], bb0 = b0p[0];

  float mu1 = 0.f, rs1 = 0.f, g1 = 0.f, bb1 = 0.f;
  if (PASS == 1) {
    const float inv1 = 1.0f / (float)(B_ * L_);
    mu1 = stats[2 + o] * inv1;
    float var1 = stats[34 + o] * inv1 - mu1 * mu1;
    rs1 = rsqrtf(var1 + EPS_);
    g1 = g1p[o]; bb1 = b1p[o];
  }

  float kk[KW_];
#pragma unroll
  for (int w = 0; w < KW_; ++w) kk[w] = kbuf[b * (OUTC * KW_) + o * KW_ + w];

  const float* xb = x + b * DENT;
  float s = 0.f, ss = 0.f;
  for (int l = threadIdx.x; l < L_; l += 128) {
    float c = 0.f;
#pragma unroll
    for (int w = 0; w < KW_; ++w) {
      float xn = (xb[l + w] - mu0) * rs0 * g0 + bb0;
      c = fmaf(xn, kk[w], c);
    }
    if (PASS == 0) { s += c; ss += c * c; }
    else {
      float hn = (c - mu1) * rs1 * g1 + bb1;
      hbf[b * HID + o * L_ + l] = f2bf(hn);
    }
  }
  if (PASS == 0) {
    __shared__ float sh[128], sh2[128];
    sh[threadIdx.x] = s; sh2[threadIdx.x] = ss; __syncthreads();
    for (int st = 64; st > 0; st >>= 1) {
      if (threadIdx.x < st) { sh[threadIdx.x] += sh[threadIdx.x + st]; sh2[threadIdx.x] += sh2[threadIdx.x + st]; }
      __syncthreads();
    }
    if (threadIdx.x == 0) { atomicAdd(&stats[2 + o], sh[0]); atomicAdd(&stats[34 + o], sh2[0]); }
  }
}

// ---------------- 5: fc_w f32 -> bf16 ----------------
__global__ __launch_bounds__(256) void k_cvt_fcw(const float* __restrict__ w, unsigned short* __restrict__ o) {
  int idx = blockIdx.x * 256 + threadIdx.x;
  if (idx < DENT * HID) o[idx] = f2bf(w[idx]);
}

// ---------------- 6: WMMA GEMM1: h[1024,400] = hbf @ fcw.T + fc_b ----------------
__global__ __launch_bounds__(256) void k_gemm1(const unsigned short* __restrict__ A,   // [1024,12544]
                                               const unsigned short* __restrict__ Bw,  // [400,12544]
                                               const float* __restrict__ bias,
                                               float* __restrict__ C) {                // [1024,400]
  const int K = HID;
  const int NT = DENT / 16;  // 25
  int wid = (blockIdx.x * 256 + threadIdx.x) >> 5;   // 1600 waves total, exact
  int lane = threadIdx.x & 31;
  int mt = wid / NT, nt = wid % NT;
  int r = lane & 15, h = lane >> 4;
  const unsigned short* pA = A + (size_t)(mt * 16 + r) * K;
  const unsigned short* pB = Bw + (size_t)(nt * 16 + r) * K;
  v8f acc = {};
  for (int kt = 0; kt < K / 32; ++kt) {
    int k0 = kt * 32;
    v8bf alo = *(const v8bf*)(pA + k0 + h * 16);
    v8bf ahi = *(const v8bf*)(pA + k0 + h * 16 + 8);
    v16bf a = __builtin_shufflevector(alo, ahi, 0, 1, 2, 3, 4, 5, 6, 7, 8, 9, 10, 11, 12, 13, 14, 15);
    v16bf b = *(const v16bf*)(pB + k0 + h * 16);
    acc = __builtin_amdgcn_wmma_f32_16x16x32_bf16(false, a, false, b, (short)0, acc, false, false);
  }
  int n = nt * 16 + r;
  float bv = bias[n];
#pragma unroll
  for (int v = 0; v < 8; ++v)
    C[(size_t)(mt * 16 + 8 * h + v) * DENT + n] = acc[v] + bv;
}

// ---------------- 7: bn2 per-feature stats ----------------
__global__ __launch_bounds__(256) void k_bn2_stats(const float* __restrict__ hf, float* stats) {
  int n = blockIdx.x;  // 400 blocks
  float s = 0.f, ss = 0.f;
  for (int b = threadIdx.x; b < B_; b += 256) {
    float v = hf[(size_t)b * DENT + n]; s += v; ss += v * v;
  }
  __shared__ float sh[256], sh2[256];
  sh[threadIdx.x] = s; sh2[threadIdx.x] = ss; __syncthreads();
  for (int st = 128; st > 0; st >>= 1) {
    if (threadIdx.x < st) { sh[threadIdx.x] += sh[threadIdx.x + st]; sh2[threadIdx.x] += sh2[threadIdx.x + st]; }
    __syncthreads();
  }
  if (threadIdx.x == 0) {
    float mu = sh[0] / (float)B_;
    float var = sh2[0] / (float)B_ - mu * mu;
    stats[66 + n] = mu;
    stats[466 + n] = rsqrtf(var + EPS_);
  }
}

// ---------------- 8: relu(bn2(h)) -> bf16, K padded to 416 ----------------
__global__ __launch_bounds__(256) void k_hr(const float* __restrict__ hf, const float* stats,
                                            const float* __restrict__ g2, const float* __restrict__ b2,
                                            unsigned short* __restrict__ hr) {
  int idx = blockIdx.x * 256 + threadIdx.x;
  if (idx >= B_ * KP) return;
  int b = idx / KP, k = idx % KP;
  unsigned short o = 0;
  if (k < DENT) {
    float v = hf[(size_t)b * DENT + k];
    float y = (v - stats[66 + k]) * stats[466 + k] * g2[k] + b2[k];
    o = f2bf(fmaxf(y, 0.f));
  }
  hr[idx] = o;
}

// ---------------- 9: all_ent_w f32 -> bf16 padded ----------------
__global__ __launch_bounds__(256) void k_cvt_ent(const float* __restrict__ w, unsigned short* __restrict__ o) {
  int idx = blockIdx.x * 256 + threadIdx.x;
  if (idx >= NENT * KP) return;
  int n = idx / KP, k = idx % KP;
  o[idx] = (k < DENT) ? f2bf(w[(size_t)n * DENT + k]) : (unsigned short)0;
}

// ---------------- 10: WMMA GEMM2: out[1024,50000] = hr @ ent.T ----------------
// Block = 8 waves. Block computes 128 M rows (8 M-tiles, one per wave) x 80 N cols
// (5 N-tiles shared by all waves). B tiles are staged in LDS in WMMA-fragment
// order so every wave reads identical fragments with stride-16B ds_load_b128
// (conflict-free). Staging uses async global->LDS copies (ASYNCcnt) when the
// toolchain exposes them. K chunked 7+6 k-steps to fit the 64KB static LDS limit.
#define G2_CHUNK 7
__global__ __launch_bounds__(256) void k_gemm2(const unsigned short* __restrict__ A,  // [1024,416]
                                               const unsigned short* __restrict__ E,  // [50000,416]
                                               float* __restrict__ Out) {             // [1024,50000]
  const int K = KP;
  const int NG = 625;                        // 3125 n-tiles / 5 per block, exact
  // LDS: per (kstep, tile j): [half(2)][lane(32)][16B] = 1KB; max 7*5 = 35KB
  __shared__ __align__(16) unsigned short ldsB[G2_CHUNK * 5 * 512];

  int mg = blockIdx.x / NG;                  // 0..7
  int ng = blockIdx.x % NG;                  // 0..624
  int lane = threadIdx.x & 31;
  int w = threadIdx.x >> 5;                  // wave id in block -> M tile
  int mt = mg * 8 + w;
  int r = lane & 15, h = lane >> 4;
  int nt0 = ng * 5;
  const unsigned short* pA = A + (size_t)(mt * 16 + r) * K;

  v8f acc[5] = {};
  for (int kb = 0; kb < K / 32; kb += G2_CHUNK) {
    int nk = (K / 32) - kb;
    if (nk > G2_CHUNK) nk = G2_CHUNK;        // 7 then 6 (block-uniform)

    // ---- stage B fragments for this k-chunk into LDS ----
    int total = nk * 5 * 64;                 // 16B granules
    for (int c = threadIdx.x; c < total; c += 256) {
      int sl   = c & 31;                     // staged lane
      int half = (c >> 5) & 1;
      int tile = c >> 6;                     // lk*5 + j
      int lk = tile / 5, j = tile - lk * 5;
      int row = (nt0 + j) * 16 + (sl & 15);
      int col = (kb + lk) * 32 + (sl >> 4) * 16 + half * 8;
      const unsigned short* gp = E + (size_t)row * K + col;
      char* lp = (char*)ldsB + (size_t)c * 16;
#if HAVE_ASYNC_LDS
      __builtin_amdgcn_global_load_async_to_lds_b128(
          (__attribute__((address_space(1))) v4i*)gp,
          (__attribute__((address_space(3))) v4i*)lp,
          0, 0);
#else
      *(uint4*)lp = *(const uint4*)gp;
#endif
    }
#if HAVE_ASYNC_LDS
    __builtin_amdgcn_s_wait_asynccnt(0);     // drain this wave's async copies
#endif
    __syncthreads();                          // make staged B visible block-wide

    // ---- compute: A from global, B fragments from LDS ----
    for (int lk = 0; lk < nk; ++lk) {
      int k0 = (kb + lk) * 32;
      v8bf alo = *(const v8bf*)(pA + k0 + h * 16);
      v8bf ahi = *(const v8bf*)(pA + k0 + h * 16 + 8);
      v16bf a = __builtin_shufflevector(alo, ahi, 0, 1, 2, 3, 4, 5, 6, 7, 8, 9, 10, 11, 12, 13, 14, 15);
#pragma unroll
      for (int j = 0; j < 5; ++j) {
        const unsigned short* bb = ldsB + (size_t)(lk * 5 + j) * 512;
        v8bf blo = *(const v8bf*)(bb + lane * 8);        // half 0: k0+h*16+[0..7]
        v8bf bhi = *(const v8bf*)(bb + 256 + lane * 8);  // half 1: k0+h*16+[8..15]
        v16bf b = __builtin_shufflevector(blo, bhi, 0, 1, 2, 3, 4, 5, 6, 7, 8, 9, 10, 11, 12, 13, 14, 15);
        acc[j] = __builtin_amdgcn_wmma_f32_16x16x32_bf16(false, a, false, b, (short)0, acc[j], false, false);
      }
    }
    __syncthreads();
  }

#pragma unroll
  for (int j = 0; j < 5; ++j) {
    int n = (nt0 + j) * 16 + r;
#pragma unroll
    for (int v = 0; v < 8; ++v)
      Out[(size_t)(mt * 16 + 8 * h + v) * NENT + n] = acc[j][v];
  }
}

extern "C" void kernel_launch(void* const* d_in, const int* in_sizes, int n_in,
                              void* d_out, int out_size, void* d_ws, size_t ws_size,
                              hipStream_t stream) {
  (void)in_sizes; (void)n_in; (void)out_size; (void)ws_size;
  const float* e1   = (const float*)d_in[0];
  const float* rel  = (const float*)d_in[1];
  const float* entw = (const float*)d_in[2];
  const float* fc1w = (const float*)d_in[3];
  const float* fc1b = (const float*)d_in[4];
  const float* fcw  = (const float*)d_in[5];
  const float* fcb  = (const float*)d_in[6];
  const float* g0   = (const float*)d_in[7];
  const float* b0   = (const float*)d_in[8];
  const float* g1   = (const float*)d_in[9];
  const float* b1   = (const float*)d_in[10];
  const float* g2   = (const float*)d_in[11];
  const float* b2   = (const float*)d_in[12];

  char* ws = (char*)d_ws;
  float*          stats  = (float*)(ws + WS_STATS);
  float*          kbuf   = (float*)(ws + WS_KBUF);
  unsigned short* hbf    = (unsigned short*)(ws + WS_HBF);
  unsigned short* fcwbf  = (unsigned short*)(ws + WS_FCWBF);
  float*          hf32   = (float*)(ws + WS_HF32);
  unsigned short* hrbf   = (unsigned short*)(ws + WS_HRBF);
  unsigned short* entbf  = (unsigned short*)(ws + WS_ENTBF);
  float*          out    = (float*)d_out;

  k_zero_stats<<<1, 256, 0, stream>>>(stats);
  k_bn0_stats<<<400, 256, 0, stream>>>(e1, stats);
  k_hyper<<<(B_ * OUTC * KW_ + 255) / 256, 256, 0, stream>>>(rel, fc1w, fc1b, kbuf);
  k_conv<0><<<B_ * OUTC, 128, 0, stream>>>(e1, kbuf, g0, b0, g1, b1, stats, hbf);
  k_conv<1><<<B_ * OUTC, 128, 0, stream>>>(e1, kbuf, g0, b0, g1, b1, stats, hbf);
  k_cvt_fcw<<<(DENT * HID + 255) / 256, 256, 0, stream>>>(fcw, fcwbf);
  k_gemm1<<<200, 256, 0, stream>>>(hbf, fcwbf, fcb, hf32);
  k_bn2_stats<<<DENT, 256, 0, stream>>>(hf32, stats);
  k_hr<<<(B_ * KP + 255) / 256, 256, 0, stream>>>(hf32, stats, g2, b2, hrbf);
  k_cvt_ent<<<(NENT * KP + 255) / 256, 256, 0, stream>>>(entw, entbf);
  k_gemm2<<<5000 /* 8 Mgroups x 625 Ngroups */, 256, 0, stream>>>(hrbf, entbf, out);
}
